// GumbelSoftmaxMessageDecoder_62706522521606
// MI455X (gfx1250) — matrix operand order repaired and checked by reading, not compiled, not run on hardware
//
#include <hip/hip_runtime.h>
#include <hip/hip_bf16.h>

// ---------------------------------------------------------------------------
// GumbelSoftmax LSTM message decoder for MI455X (gfx1250, wave32, WMMA).
//
// bf16 operands -> v_wmma_f32_16x16x32_bf16, f32 accumulate, f32 elementwise.
// Embedding matmul folded into gates GEMM (W_combo = W_emb @ W_ih^T).
// Weights pre-swizzled into WMMA-fragment order so B fragments are coalesced
// 32B/lane global b128 loads (served from L2/WGP$). A tiles are double-
// buffered into LDS via GLOBAL_LOAD_ASYNC_TO_LDS_B128 (no VGPR round-trip),
// fenced with s_wait_asynccnt, and consumed as contiguous ds_load_b128
// fragments.
// ---------------------------------------------------------------------------

typedef __attribute__((ext_vector_type(16))) __bf16 v16bf;
typedef __attribute__((ext_vector_type(8)))  float  v8f;

#define B_    16384
#define D_    512
#define A_    65
#define T_    16
#define KIN   608            // 96 (padded onehot) + 512 (h)
#define NKIN  19             // KIN/32 k-steps
#define NG4   2048           // 4*D
#define NPAD  96             // padded action width
#define MSGLEN (17 * 65)
#define ASTR  40             // padded LDS row stride (bf16) for A tiles
#define FRAG  512            // elements per swizzled B fragment (32 lanes * 16)

static __device__ __forceinline__ float sigmoidf_(float x) {
  return 1.0f / (1.0f + expf(-x));
}

static __device__ __forceinline__ v8f zero8() {
  v8f z = {0.f, 0.f, 0.f, 0.f, 0.f, 0.f, 0.f, 0.f};
  return z;
}

// Async 16B/lane global->LDS copy (CDNA5 TDM-adjacent async path, ASYNCcnt).
// Low 32 bits of a generic __shared__ pointer are the LDS byte offset.
static __device__ __forceinline__ void async_copy16(void* lds_dst, const void* gsrc) {
  unsigned lds_off = (unsigned)(unsigned long long)(uintptr_t)lds_dst;
  asm volatile("global_load_async_to_lds_b128 %0, %1, off"
               :
               : "v"(lds_off), "v"(gsrc)
               : "memory");
}

static __device__ __forceinline__ void wait_async0() {
  asm volatile("s_wait_asynccnt 0x0" ::: "memory");
}

// A fragment from LDS row-major tile (stride ASTR). Each lane = two 16B runs:
// a[0..7] = As[m][hf*16 .. +7], a[8..15] = As[m][hf*16+8 .. +7]
static __device__ __forceinline__ v16bf frag_a_lds(const __bf16* As, int mt, int lane) {
  const int m  = (mt << 4) + (lane & 15);
  const int hf = lane >> 4;
  union { uint4 u[2]; v16bf v; } r;
  r.u[0] = *(const uint4*)(As + m * ASTR + hf * 16);
  r.u[1] = *(const uint4*)(As + m * ASTR + hf * 16 + 8);
  return r.v;
}

// B fragment from globally pre-swizzled weights: [frag][lane][16] bf16.
static __device__ __forceinline__ v16bf frag_b_glob(const __bf16* W, int frag, int lane) {
  union { uint4 u[2]; v16bf v; } r;
  const uint4* p = (const uint4*)(W + ((size_t)frag * 32 + lane) * 16);
  r.u[0] = p[0];
  r.u[1] = p[1];
  return r.v;
}

static __device__ __forceinline__ v8f wmma_bf16(v16bf a, v16bf b, v8f c) {
  return __builtin_amdgcn_wmma_f32_16x16x32_bf16(false, a, false, b, (short)0, c,
                                                 false, false);
}

// ------------------------- one-time prep kernels ---------------------------
// Swizzled gates weights WgS: [19 kk][128 ntile][32 lane][16 e]
//   k = kk*32 + (lane>>4)*16 + e  (row of [onehot|h] input, 0..607)
//   n = ntile*16 + (lane&15)      (gate output col, 0..2047)
// rows 0..66 = W_emb @ W_ih^T (embedding folded), 67..95 = 0, 96.. = W_hh^T
__global__ void prep_wg(const float* __restrict__ W_emb,
                        const float* __restrict__ W_ih,
                        const float* __restrict__ W_hh,
                        __bf16* __restrict__ WgS) {
  int idx = blockIdx.x * blockDim.x + threadIdx.x;
  if (idx >= NKIN * 128 * FRAG) return;
  const int e = idx & 15, lane = (idx >> 4) & 31;
  const int ntg = (idx >> 9) & 127, kk = idx >> 16;
  const int k = kk * 32 + ((lane >> 4) << 4) + e;
  const int n = (ntg << 4) + (lane & 15);
  float v = 0.f;
  if (k < NPAD) {
    if (k < 67) {
      const float* em = W_emb + (size_t)k * D_;
      const float* wi = W_ih + (size_t)n * D_;
#pragma unroll 8
      for (int d = 0; d < D_; ++d) v += em[d] * wi[d];
    }
  } else {
    v = W_hh[(size_t)n * D_ + (k - NPAD)];
  }
  WgS[idx] = (__bf16)v;
}

// Swizzled W_act^T: [16 kk][6 ntile][32 lane][16 e]; n>=65 zero-padded.
__global__ void prep_wact(const float* __restrict__ W_act, __bf16* __restrict__ WactS) {
  int idx = blockIdx.x * blockDim.x + threadIdx.x;
  if (idx >= 16 * 6 * FRAG) return;
  const int e = idx & 15, lane = (idx >> 4) & 31;
  const int nt = (idx >> 9) % 6, kk = idx / (6 * FRAG);
  const int k = kk * 32 + ((lane >> 4) << 4) + e;
  const int n = (nt << 4) + (lane & 15);
  WactS[idx] = (n < A_) ? (__bf16)W_act[(size_t)n * D_ + k] : (__bf16)0.f;
}

// Swizzled W_cell^T / W_hid^T: [16 kk][32 ntile][32 lane][16 e]
__global__ void prep_initw(const float* __restrict__ W_cell,
                           const float* __restrict__ W_hid,
                           __bf16* __restrict__ WcS, __bf16* __restrict__ WhS) {
  int idx = blockIdx.x * blockDim.x + threadIdx.x;
  if (idx >= 16 * 32 * FRAG) return;
  const int e = idx & 15, lane = (idx >> 4) & 31;
  const int nt = (idx >> 9) & 31, kk = idx >> 14;
  const int k = kk * 32 + ((lane >> 4) << 4) + e;
  const int n = (nt << 4) + (lane & 15);
  WcS[idx] = (__bf16)W_cell[(size_t)n * D_ + k];
  WhS[idx] = (__bf16)W_hid[(size_t)n * D_ + k];
}

__global__ void prep_bias(const float* __restrict__ b_ih,
                          const float* __restrict__ b_hh,
                          float* __restrict__ bias) {
  int n = blockIdx.x * blockDim.x + threadIdx.x;
  if (n < NG4) bias[n] = b_ih[n] + b_hh[n];
}

// cols 0..95 of step-0 input: one-hot at BOS (col 66)
__global__ void prep_xin0(__bf16* __restrict__ xin) {
  int idx = blockIdx.x * blockDim.x + threadIdx.x;
  if (idx >= B_ * NPAD) return;
  int b = idx / NPAD, j = idx % NPAD;
  xin[(size_t)b * KIN + j] = (__bf16)((j == 66) ? 1.f : 0.f);
}

// ------------------------ init projections (WMMA) --------------------------
__global__ __launch_bounds__(256) void init_proj(
    const float* __restrict__ encoded,          // [B][512] f32
    const __bf16* __restrict__ WcS,             // swizzled [16][32][32][16]
    const __bf16* __restrict__ WhS,
    const float* __restrict__ b_cell, const float* __restrict__ b_hid,
    float* __restrict__ cbuf,                   // [B][512]
    __bf16* __restrict__ xout) {                // [B][608]
  __shared__ __bf16 As[2][64 * ASTR];
  const int tid = threadIdx.x, wave = tid >> 5, lane = tid & 31;
  const int mt = wave & 3, npair = wave >> 2;
  const int bm0 = blockIdx.x * 64, n0 = blockIdx.y * 64, nt0 = n0 >> 4;
  const int r = tid >> 2, s = (tid & 3) * 8;
  v8f acc[2][2];
#pragma unroll
  for (int g = 0; g < 2; ++g)
#pragma unroll
    for (int j = 0; j < 2; ++j) acc[g][j] = zero8();

  auto stage = [&](int kk) {   // f32 -> bf16 conversion in flight
    const float4* p = (const float4*)(encoded + (size_t)(bm0 + r) * D_ + kk * 32 + s);
    float4 x0 = p[0], x1 = p[1];
    __bf16* dst = &As[kk & 1][r * ASTR + s];
    dst[0] = (__bf16)x0.x; dst[1] = (__bf16)x0.y;
    dst[2] = (__bf16)x0.z; dst[3] = (__bf16)x0.w;
    dst[4] = (__bf16)x1.x; dst[5] = (__bf16)x1.y;
    dst[6] = (__bf16)x1.z; dst[7] = (__bf16)x1.w;
  };
  stage(0);
  for (int kk = 0; kk < 16; ++kk) {
    __syncthreads();
    if (kk + 1 < 16) stage(kk + 1);
    v16bf a = frag_a_lds(&As[kk & 1][0], mt, lane);
    const int fbase = kk * 32 + nt0 + npair * 2;
#pragma unroll
    for (int j = 0; j < 2; ++j) {
      v16bf b0 = frag_b_glob(WcS, fbase + j, lane);
      acc[0][j] = wmma_bf16(a, b0, acc[0][j]);
      v16bf b1 = frag_b_glob(WhS, fbase + j, lane);
      acc[1][j] = wmma_bf16(a, b1, acc[1][j]);
    }
  }
  const int hf = lane >> 4, ln = lane & 15;
#pragma unroll
  for (int j = 0; j < 2; ++j) {
    const int d = n0 + (npair * 2 + j) * 16 + ln;
    const float bc = b_cell[d], bh = b_hid[d];
#pragma unroll
    for (int e = 0; e < 8; ++e) {
      const int b = bm0 + mt * 16 + hf * 8 + e;
      xout[(size_t)b * KIN + NPAD + d] = (__bf16)(acc[0][j][e] + bc);
      cbuf[(size_t)b * D_ + d] = acc[1][j][e] + bh;
    }
  }
}

// ------------------------ per-step gates + cell ----------------------------
__global__ __launch_bounds__(256) void lstm_step(
    const __bf16* __restrict__ xin,   // [B][608] (onehot_t | h_{t-1})
    const __bf16* __restrict__ WgS,   // swizzled [19][128][32][16]
    const float* __restrict__ bias,   // [2048]
    float* __restrict__ cbuf,         // [B][512] in/out
    __bf16* __restrict__ xout) {      // [B][608], writes cols 96..607 (h_t)
  __shared__ __bf16 As[2][64 * ASTR];
  const int tid = threadIdx.x, wave = tid >> 5, lane = tid & 31;
  const int mt = wave & 3, npair = wave >> 2;
  const int bm0 = blockIdx.x * 64, n0 = blockIdx.y * 64, nt0 = n0 >> 4;
  const int r = tid >> 2, s = (tid & 3) * 8;
  v8f acc[4][2];
#pragma unroll
  for (int g = 0; g < 4; ++g)
#pragma unroll
    for (int j = 0; j < 2; ++j) acc[g][j] = zero8();

  auto stage = [&](int kk) {  // async global->LDS, no VGPR round-trip
    async_copy16(&As[kk & 1][r * ASTR + s],
                 xin + (size_t)(bm0 + r) * KIN + kk * 32 + s);
  };
  stage(0);
  for (int kk = 0; kk < NKIN; ++kk) {
    wait_async0();            // staged tile for this kk has landed in LDS
    __syncthreads();
    if (kk + 1 < NKIN) {
      stage(kk + 1);
      // warm next k-step's weight block (hits L2/WGP$)
      __builtin_prefetch(WgS + (size_t)(kk + 1) * 128 * FRAG +
                             ((size_t)(nt0 + npair * 2) * 32 + lane) * 16,
                         0, 1);
    }
    v16bf a = frag_a_lds(&As[kk & 1][0], mt, lane);
    const __bf16* wk = WgS + (size_t)kk * 128 * FRAG;
#pragma unroll
    for (int g = 0; g < 4; ++g)
#pragma unroll
      for (int j = 0; j < 2; ++j) {
        v16bf b = frag_b_glob(wk, g * 32 + nt0 + npair * 2 + j, lane);
        acc[g][j] = wmma_bf16(a, b, acc[g][j]);
      }
  }
  // fused LSTM cell: i,f,g,o -> c_t, h_t
  const int hf = lane >> 4, ln = lane & 15;
#pragma unroll
  for (int j = 0; j < 2; ++j) {
    const int d = n0 + (npair * 2 + j) * 16 + ln;
    const float bi = bias[d], bf = bias[D_ + d], bg = bias[2 * D_ + d],
                bo = bias[3 * D_ + d];
#pragma unroll
    for (int e = 0; e < 8; ++e) {
      const int b = bm0 + mt * 16 + hf * 8 + e;
      const float iv = sigmoidf_(acc[0][j][e] + bi);
      const float fv = sigmoidf_(acc[1][j][e] + bf);
      const float gv = tanhf(acc[2][j][e] + bg);
      const float ov = sigmoidf_(acc[3][j][e] + bo);
      const size_t ci = (size_t)b * D_ + d;
      const float cn = fv * cbuf[ci] + iv * gv;
      cbuf[ci] = cn;
      xout[(size_t)b * KIN + NPAD + d] = (__bf16)(ov * tanhf(cn));
    }
  }
}

// -------------------- per-step logits + gumbel softmax ---------------------
__global__ __launch_bounds__(256) void logits_softmax(
    __bf16* __restrict__ xin,              // [B][608]; read h (96..), write action (0..95)
    const __bf16* __restrict__ WactS,      // swizzled [16][6][32][16]
    const float* __restrict__ b_act,       // [65]
    const float* __restrict__ gumbel_u,    // [16][B][65]
    int t,
    float* __restrict__ out) {             // message [B][17][65] ++ eos [B][17]
  __shared__ __bf16 As[2][64 * ASTR];
  __shared__ float  Lg[64 * 96];
  const int tid = threadIdx.x, wave = tid >> 5, lane = tid & 31;
  const int mt = wave & 3, g3 = wave >> 2;  // n-tiles 3*g3 .. 3*g3+2
  const int bm0 = blockIdx.x * 64;
  const int r = tid >> 2, s = (tid & 3) * 8;
  v8f acc[3];
#pragma unroll
  for (int j = 0; j < 3; ++j) acc[j] = zero8();

  auto stage = [&](int kk) {
    async_copy16(&As[kk & 1][r * ASTR + s],
                 xin + (size_t)(bm0 + r) * KIN + NPAD + kk * 32 + s);
  };
  stage(0);
  for (int kk = 0; kk < 16; ++kk) {
    wait_async0();
    __syncthreads();
    if (kk + 1 < 16) stage(kk + 1);
    v16bf a = frag_a_lds(&As[kk & 1][0], mt, lane);
#pragma unroll
    for (int j = 0; j < 3; ++j) {
      v16bf b = frag_b_glob(WactS, kk * 6 + 3 * g3 + j, lane);
      acc[j] = wmma_bf16(a, b, acc[j]);
    }
  }
  // spill logits to LDS for row-wise softmax
  const int hf = lane >> 4, ln = lane & 15;
  __syncthreads();
#pragma unroll
  for (int j = 0; j < 3; ++j)
#pragma unroll
    for (int e = 0; e < 8; ++e)
      Lg[(mt * 16 + hf * 8 + e) * 96 + (3 * g3 + j) * 16 + ln] = acc[j][e];
  __syncthreads();

  if (tid < 64) {
    const int rr = tid;
    const size_t b = bm0 + rr;
    float* L = Lg + rr * 96;
    float m1 = -1e30f;
    for (int a = 0; a < A_; ++a) {
      L[a] += b_act[a];
      m1 = fmaxf(m1, L[a]);
    }
    float se = 0.f;
    for (int a = 0; a < A_; ++a) se += expf(L[a] - m1);
    const float lse = m1 + logf(se);
    const float eosp = expf(L[0] - lse);            // EOS == action 0
    const float* gu = gumbel_u + ((size_t)t * B_ + b) * A_;
    float m2 = -1e30f;
    for (int a = 0; a < A_; ++a) {
      const float sc = (L[a] - lse) - logf(-logf(gu[a]));  // logp + gumbel, TEMP=1
      L[a] = sc;
      m2 = fmaxf(m2, sc);
    }
    float s2 = 0.f;
    for (int a = 0; a < A_; ++a) s2 += expf(L[a] - m2);
    const float inv = 1.f / s2;
    float* msg = out + b * MSGLEN + (size_t)t * A_;
    __bf16* xr = xin + b * (size_t)KIN;
    for (int a = 0; a < A_; ++a) {
      const float act = expf(L[a] - m2) * inv;
      msg[a] = act;
      xr[a] = (__bf16)act;                           // next-step onehot
    }
    for (int a = A_; a < NPAD; ++a) xr[a] = (__bf16)0.f;
    out[(size_t)B_ * MSGLEN + b * 17 + t] = eosp;
  }
}

// ----------------------------- finalize ------------------------------------
__global__ void finalize(float* __restrict__ out) {
  int idx = blockIdx.x * blockDim.x + threadIdx.x;
  if (idx >= B_ * A_) return;
  const size_t b = idx / A_;
  const int a = idx % A_;
  out[b * MSGLEN + (size_t)T_ * A_ + a] = (a == 0) ? 1.f : 0.f;
  if (a == 0) out[(size_t)B_ * MSGLEN + b * 17 + T_] = 1.f;
}

// ----------------------------- launcher ------------------------------------
extern "C" void kernel_launch(void* const* d_in, const int* in_sizes, int n_in,
                              void* d_out, int out_size, void* d_ws, size_t ws_size,
                              hipStream_t stream) {
  (void)in_sizes; (void)n_in; (void)out_size; (void)ws_size;
  const float* encoded  = (const float*)d_in[0];
  const float* gumbel_u = (const float*)d_in[1];
  const float* W_emb    = (const float*)d_in[2];
  const float* W_cell   = (const float*)d_in[3];
  const float* b_cell   = (const float*)d_in[4];
  const float* W_hid    = (const float*)d_in[5];
  const float* b_hid    = (const float*)d_in[6];
  const float* W_ih     = (const float*)d_in[7];
  const float* W_hh     = (const float*)d_in[8];
  const float* b_ih     = (const float*)d_in[9];
  const float* b_hh     = (const float*)d_in[10];
  const float* W_act    = (const float*)d_in[11];
  const float* b_act    = (const float*)d_in[12];
  float* out = (float*)d_out;

  char* ws = (char*)d_ws;
  size_t o = 0;
  auto take = [&](size_t bytes) {
    char* p = ws + o;
    o = (o + bytes + 255) & ~(size_t)255;
    return p;
  };
  __bf16* WgS   = (__bf16*)take((size_t)NKIN * 128 * FRAG * 2);  // 2.49 MB
  float*  bias  = (float*) take((size_t)NG4 * 4);
  __bf16* WactS = (__bf16*)take((size_t)16 * 6 * FRAG * 2);
  __bf16* WcS   = (__bf16*)take((size_t)16 * 32 * FRAG * 2);
  __bf16* WhS   = (__bf16*)take((size_t)16 * 32 * FRAG * 2);
  float*  cbuf  = (float*) take((size_t)B_ * D_ * 4);            // 33.5 MB
  __bf16* xinA  = (__bf16*)take((size_t)B_ * KIN * 2);           // 19.9 MB
  __bf16* xinB  = (__bf16*)take((size_t)B_ * KIN * 2);           // 19.9 MB

  prep_wg<<<(NKIN * 128 * FRAG + 255) / 256, 256, 0, stream>>>(W_emb, W_ih, W_hh, WgS);
  prep_wact<<<(16 * 6 * FRAG + 255) / 256, 256, 0, stream>>>(W_act, WactS);
  prep_initw<<<(16 * 32 * FRAG + 255) / 256, 256, 0, stream>>>(W_cell, W_hid, WcS, WhS);
  prep_bias<<<(NG4 + 255) / 256, 256, 0, stream>>>(b_ih, b_hh, bias);
  prep_xin0<<<(B_ * NPAD + 255) / 256, 256, 0, stream>>>(xinA);

  init_proj<<<dim3(B_ / 64, D_ / 64), 256, 0, stream>>>(encoded, WcS, WhS,
                                                        b_cell, b_hid, cbuf, xinA);

  __bf16* cur = xinA;
  __bf16* nxt = xinB;
  for (int t = 0; t < T_; ++t) {
    lstm_step<<<dim3(B_ / 64, D_ / 64), 256, 0, stream>>>(cur, WgS, bias, cbuf, nxt);
    logits_softmax<<<B_ / 64, 256, 0, stream>>>(nxt, WactS, b_act, gumbel_u, t, out);
    __bf16* tmp = cur; cur = nxt; nxt = tmp;
  }
  finalize<<<(B_ * A_ + 255) / 256, 256, 0, stream>>>(out);
}
